// SelfAtt_65524021067849
// MI455X (gfx1250) — compile-verified, matrix operand order checked by reading, not run on hardware
//
#include <hip/hip_runtime.h>

// ---------------------------------------------------------------------------
// Self-attention block (GroupNorm -> qkv GEMM -> softmax-k linear attention
// -> proj GEMM) for MI455X / gfx1250, wave32, WMMA f32_16x16x32_f16,
// TDM (tensor_load_to_lds) staging of K/V head panels in the attention stage.
// ---------------------------------------------------------------------------

typedef _Float16 half_t;
typedef _Float16 v16h __attribute__((ext_vector_type(16)));
typedef float    v8f  __attribute__((ext_vector_type(8)));
typedef unsigned uint4v __attribute__((ext_vector_type(4)));
typedef int      int8v  __attribute__((ext_vector_type(8)));
typedef int      int4v  __attribute__((ext_vector_type(4)));

#if defined(__has_builtin)
#  if __has_builtin(__builtin_amdgcn_tensor_load_to_lds) && \
      __has_builtin(__builtin_amdgcn_s_wait_tensorcnt)
#    define USE_TDM 1
#  endif
#endif
#ifndef USE_TDM
#  define USE_TDM 0
#endif

constexpr int BB  = 16;    // batch
constexpr int CC  = 256;   // channels
constexpr int NN  = 4096;  // h*w
constexpr int HH  = 8;     // heads
constexpr int NG  = 32;    // norm groups
constexpr int CPG = 8;     // channels per group
constexpr int OC  = 768;   // 3*C

// workspace layout (bytes)
constexpr size_t OFF_STATS = 0;                                   // 512*2*4
constexpr size_t OFF_WQ    = 4096;                                // 768*256 f16
constexpr size_t OFF_WP    = OFF_WQ + (size_t)OC * CC * 2;        // 256*256 f16
constexpr size_t OFF_XNT   = OFF_WP + (size_t)CC * CC * 2;        // [b][n][c] f16
constexpr size_t OFF_QKVT  = OFF_XNT + (size_t)BB * NN * CC * 2;  // [b][n][768] f16
constexpr size_t OFF_OUTT  = OFF_QKVT + (size_t)BB * NN * OC * 2; // [b][n][c] f16

// A/B fragment K-base for v_wmma_f32_16x16x32_f16 (ISA 7.12.2, 16-bit A 16x32):
// VGPR v in 0..3 -> K = 2v + 8*laneHalf ; v in 4..7 -> K = 16 + 2(v-4) + 8*laneHalf
__device__ __forceinline__ int kbase(int v, int lhalf) {
    return ((v < 4) ? (2 * v) : (16 + 2 * (v - 4))) + 8 * lhalf;
}

__device__ __forceinline__ unsigned pack2h(float a, float b) {
    union { _Float16 h[2]; unsigned u; } p;
    p.h[0] = (_Float16)a; p.h[1] = (_Float16)b;
    return p.u;
}

__device__ __forceinline__ v8f wmma16(v16h a, v16h b, v8f c) {
    return __builtin_amdgcn_wmma_f32_16x16x32_f16(
        /*neg_a=*/false, a, /*neg_b=*/false, b,
        /*c_mod=*/(short)0, c, /*reuse_a=*/false, /*reuse_b=*/false);
}

__device__ __forceinline__ v8f zero8() {
    v8f z = {0.f, 0.f, 0.f, 0.f, 0.f, 0.f, 0.f, 0.f};
    return z;
}

union FragH { unsigned u[8]; v16h h; };

#if USE_TDM
// Issue one 2-D TDM load: tile (tile_d1 rows x tile_d0 elems) of 16-bit data,
// row stride `stride0` elements, into LDS at byte offset lds_off.
// LDS padding: +1 dword after every 64B row (pad_enable, interval=64B, amount=1dw)
// so the landed row pitch is 68 bytes (34 halves).
// 6-arg builtin form (amdgpu-toolchain / clang-23): groups 2/3 and the extra
// descriptor vector are unused for 2-D tensors -> zero-filled.
__device__ __forceinline__ void tdm_load_2d(unsigned lds_off, const void* gaddr,
                                            unsigned tensor_d0, unsigned tensor_d1,
                                            unsigned tile_d0, unsigned tile_d1,
                                            unsigned stride0) {
    unsigned long long ga = (unsigned long long)(uintptr_t)gaddr;
    uint4v g0;
    g0.x = 1u;                                        // count=1, user mode
    g0.y = lds_off;                                   // lds_addr
    g0.z = (unsigned)(ga & 0xFFFFFFFFu);              // global_addr[31:0]
    g0.w = (unsigned)((ga >> 32) & 0x01FFFFFFu)       // global_addr[56:32]
         | (2u << 30);                                // type = 2 (image)
    int8v g1;
    g1[0] = (int)((1u << 16)                          // data_size = 1 -> 2 bytes
                | (1u << 20)                          // pad_enable
                | (3u << 22));                        // pad_interval = 3 -> 64B
    g1[1] = (int)((tensor_d0 & 0xFFFFu) << 16);       // tensor_dim0[15:0]
    g1[2] = (int)(((tensor_d0 >> 16) & 0xFFFFu)
                | ((tensor_d1 & 0xFFFFu) << 16));     // dim0 hi | dim1 lo
    g1[3] = (int)(((tensor_d1 >> 16) & 0xFFFFu)
                | ((tile_d0 & 0xFFFFu) << 16));       // dim1 hi | tile0
    g1[4] = (int)(tile_d1 & 0xFFFFu);                 // tile1 | tile2=0
    g1[5] = (int)stride0;                             // tensor_dim0_stride lo
    g1[6] = 0;                                        // stride0 hi | stride1 lo
    g1[7] = 0;                                        // stride1 hi
    int4v gz4 = {0, 0, 0, 0};
    int8v gz8 = {0, 0, 0, 0, 0, 0, 0, 0};
    __builtin_amdgcn_tensor_load_to_lds(g0, g1, gz4, gz4, gz8, 0);
}
#endif

// ---------------------------------------------------------------------------
// Kernel 0: convert weights to f16 (row-major [o][c], so B-pairs over k=c are
// contiguous dwords).
// ---------------------------------------------------------------------------
__global__ void k_prep_w(const float* __restrict__ wqkv,
                         const float* __restrict__ wproj,
                         half_t* __restrict__ wq_h,
                         half_t* __restrict__ wp_h) {
    int i = blockIdx.x * 256 + threadIdx.x;
    if (i < OC * CC) wq_h[i] = (half_t)wqkv[i];
    if (i < CC * CC) wp_h[i] = (half_t)wproj[i];
}

// ---------------------------------------------------------------------------
// Kernel 1: GroupNorm statistics. One block per (b, group); 8 ch * 4096 px
// contiguous -> fully coalesced reduction.
// ---------------------------------------------------------------------------
__global__ __launch_bounds__(256) void k_gn_stats(const float* __restrict__ x,
                                                  float* __restrict__ stats) {
    int bg = blockIdx.x;                              // 0..511
    const float* p = x + (size_t)bg * (CPG * NN);
    float s = 0.f, sq = 0.f;
    for (int i = threadIdx.x; i < CPG * NN; i += 256) {
        float v = p[i];
        s += v; sq += v * v;
    }
    __shared__ float ls[256], lq[256];
    ls[threadIdx.x] = s; lq[threadIdx.x] = sq;
    __syncthreads();
    for (int st = 128; st > 0; st >>= 1) {
        if (threadIdx.x < (unsigned)st) {
            ls[threadIdx.x] += ls[threadIdx.x + st];
            lq[threadIdx.x] += lq[threadIdx.x + st];
        }
        __syncthreads();
    }
    if (threadIdx.x == 0) {
        float inv = 1.f / (float)(CPG * NN);
        float m   = ls[0] * inv;
        float var = lq[0] * inv - m * m;
        stats[bg * 2 + 0] = m;
        stats[bg * 2 + 1] = rsqrtf(var + 1e-5f);
    }
}

// ---------------------------------------------------------------------------
// Kernel 2: apply GN + affine, transpose to pixel-major xn_t[b][n][c] (f16)
// through LDS so the 512B output rows store coalesced.
// ---------------------------------------------------------------------------
__global__ __launch_bounds__(256) void k_gn_apply(const float* __restrict__ x,
                                                  const float* __restrict__ stats,
                                                  const float* __restrict__ gw,
                                                  const float* __restrict__ gb,
                                                  half_t* __restrict__ xnt) {
    int b  = blockIdx.x >> 6;       // /64
    int nt = blockIdx.x & 63;
    int n0 = nt * 64;
    __shared__ _Float16 tile[CC][66];     // pad
    __shared__ float sm[NG], sr[NG];
    int tid = threadIdx.x;
    if (tid < NG) {
        sm[tid] = stats[(b * NG + tid) * 2 + 0];
        sr[tid] = stats[(b * NG + tid) * 2 + 1];
    }
    __syncthreads();
    int q  = tid >> 6;      // 0..3
    int nl = tid & 63;
    for (int i = 0; i < 64; ++i) {
        int c = i * 4 + q;
        float v = x[((size_t)(b * CC + c)) * NN + n0 + nl];
        int g = c >> 3;
        float y = (v - sm[g]) * sr[g] * gw[c] + gb[c];
        tile[c][nl] = (_Float16)y;
    }
    __syncthreads();
    int dw   = tid & 127;   // dword column (2 channels)
    int rsel = tid >> 7;    // 0..1
    unsigned* dst = (unsigned*)xnt;
    for (int j = 0; j < 32; ++j) {
        int nl2 = j * 2 + rsel;
        union { _Float16 h[2]; unsigned u; } pk;
        pk.h[0] = tile[2 * dw + 0][nl2];
        pk.h[1] = tile[2 * dw + 1][nl2];
        dst[((size_t)(b * NN) + n0 + nl2) * (CC / 2) + dw] = pk.u;
    }
}

// ---------------------------------------------------------------------------
// Kernel 3: qkv GEMM.  qkv_t[b][n][o] = sum_c xn_t[b][n][c] * Wqkv[o][c].
// Block = one 16-row n-tile; 8 waves * 6 o-tiles each = 48 o-tiles (768).
// A pairs contiguous in xn_t rows, B pairs contiguous in Wqkv rows.
// ---------------------------------------------------------------------------
__global__ __launch_bounds__(256) void k_qkv(const half_t* __restrict__ xnt,
                                             const half_t* __restrict__ wq,
                                             half_t* __restrict__ qkvt) {
    int b    = blockIdx.x >> 8;
    int mt   = blockIdx.x & 255;
    int n0   = mt * 16;
    int tid  = threadIdx.x;
    int wave = tid >> 5;
    int lane = tid & 31;
    int col  = lane & 15;
    int lhf  = lane >> 4;

    const half_t* arow = xnt + ((size_t)(b * NN) + n0 + col) * CC;
    const half_t* brow[6];
#pragma unroll
    for (int t = 0; t < 6; ++t) {
        int o = (wave * 6 + t) * 16 + col;
        brow[t] = wq + (size_t)o * CC;
    }

    v8f acc[6];
#pragma unroll
    for (int t = 0; t < 6; ++t) acc[t] = zero8();

    for (int ks = 0; ks < 8; ++ks) {
        int k0 = ks * 32;
        __builtin_prefetch(arow + k0 + 32, 0, 0);
        FragH A;
#pragma unroll
        for (int v = 0; v < 8; ++v)
            A.u[v] = *(const unsigned*)(arow + k0 + kbase(v, lhf));
#pragma unroll
        for (int t = 0; t < 6; ++t) {
            FragH Bf;
#pragma unroll
            for (int v = 0; v < 8; ++v)
                Bf.u[v] = *(const unsigned*)(brow[t] + k0 + kbase(v, lhf));
            acc[t] = wmma16(A.h, Bf.h, acc[t]);
        }
    }

    // per-wave LDS transpose -> pixel-major f16 stores (16B per lane)
    __shared__ float xch[8][16][17];
    for (int t = 0; t < 6; ++t) {
#pragma unroll
        for (int r = 0; r < 8; ++r)
            xch[wave][r + 8 * lhf][col] = acc[t][r];
        const float* srcp = &xch[wave][col][lhf * 8];
        union { _Float16 h[8]; uint4 u; } pk;
#pragma unroll
        for (int e = 0; e < 8; ++e) pk.h[e] = (_Float16)srcp[e];
        *(uint4*)(qkvt + ((size_t)(b * NN) + n0 + col) * OC +
                  (wave * 6 + t) * 16 + lhf * 8) = pk.u;
    }
}

// ---------------------------------------------------------------------------
// Kernel 4: per-(b,head) attention. qkv_t is L2-resident (96MB < 192MB).
//   1) row max / sum-exp of k over n (softmax stats, fp32)
//   2) att[d][e] = sum_n exp(k[d][n]-Md) * v[e][n]   (WMMA, K=4096; K/V head
//      panels staged to LDS by the Tensor Data Mover, packed to pair layout,
//      8 waves = 2 k-buffers x 4 (d,e) tile combos; partials via ds_add_f32)
//   3) att[d][e] /= Zd ; pack as B-fragment layout in LDS
//   4) out_t[n][e] = sum_d q_t[n][d] * att[d][e]     (WMMA, K=32, A from L2)
// ---------------------------------------------------------------------------
__global__ __launch_bounds__(256) void k_att(const half_t* __restrict__ qkvt,
                                             half_t* __restrict__ outt) {
    int b    = blockIdx.x >> 3;
    int h    = blockIdx.x & 7;
    int tid  = threadIdx.x;
    int wave = tid >> 5;
    int lane = tid & 31;
    int col  = lane & 15;
    int lhf  = lane >> 4;

    const int kcol0 = 256 + h * 32;
    const int vcol0 = 512 + h * 32;
    const int qcol0 = h * 32;
    const half_t* base = qkvt + (size_t)b * NN * OC;

    __shared__ float red[8][32];
    __shared__ float Md[32], Zd[32];

    // ---- softmax statistics over n for each of the 32 d-rows ----
    {
        int d = tid & 31, ch = tid >> 5;          // 8 chunks of 512 pixels
        float m = -1e30f;
        for (int n = ch * 512; n < ch * 512 + 512; ++n)
            m = fmaxf(m, (float)base[(size_t)n * OC + kcol0 + d]);
        red[ch][d] = m;
        __syncthreads();
        if (tid < 32) {
            float mm = red[0][tid];
#pragma unroll
            for (int i = 1; i < 8; ++i) mm = fmaxf(mm, red[i][tid]);
            Md[tid] = mm;
        }
        __syncthreads();
        float s = 0.f;
        for (int n = ch * 512; n < ch * 512 + 512; ++n)
            s += __expf((float)base[(size_t)n * OC + kcol0 + d] - Md[d]);
        red[ch][d] = s;
        __syncthreads();
        if (tid < 32) {
            float ss = 0.f;
#pragma unroll
            for (int i = 0; i < 8; ++i) ss += red[i][tid];
            Zd[tid] = ss;
        }
        __syncthreads();
    }

    // ---- att = exp(k - Md) @ v^T  (32x32, K = 4096) ----
    __shared__ unsigned Apack[2][32][17];   // [kbuf][d][n-pair]
    __shared__ unsigned Bpack[2][32][17];   // [kbuf][e][n-pair]
    __shared__ float attM[32][33];
#if USE_TDM
    // TDM-landed raw tiles: 32 rows x 32 halves, row pitch 34 halves (68B, padded)
    __shared__ __align__(16) _Float16 rawK[2][32][34];
    __shared__ __align__(16) _Float16 rawV[2][32][34];
    const unsigned ldsK0 = (unsigned)(uintptr_t)&rawK[0][0][0];
    const unsigned ldsK1 = (unsigned)(uintptr_t)&rawK[1][0][0];
    const unsigned ldsV0 = (unsigned)(uintptr_t)&rawV[0][0][0];
    const unsigned ldsV1 = (unsigned)(uintptr_t)&rawV[1][0][0];
#endif

    for (int i = tid; i < 32 * 32; i += 256) attM[i >> 5][i & 31] = 0.f;
    __syncthreads();

    const int kb_dt = (wave >> 1) & 1;      // d-tile for this wave
    const int kb_et = wave & 1;             // e-tile
    const int kb_bf = wave >> 2;            // which k-buffer this wave consumes
    v8f accA = zero8();

    for (int it = 0; it < 64; ++it) {
#if USE_TDM
        // -- DMA both k-buffers' K and V panels into LDS (one wave issues) --
        if (wave == 0) {
            int nA = (it * 2 + 0) * 32;
            int nB = (it * 2 + 1) * 32;
            tdm_load_2d(ldsK0, base + (size_t)nA * OC + kcol0, 32, 32, 32, 32, OC);
            tdm_load_2d(ldsK1, base + (size_t)nB * OC + kcol0, 32, 32, 32, 32, OC);
            tdm_load_2d(ldsV0, base + (size_t)nA * OC + vcol0, 32, 32, 32, 32, OC);
            tdm_load_2d(ldsV1, base + (size_t)nB * OC + vcol0, 32, 32, 32, 32, OC);
            __builtin_amdgcn_s_wait_tensorcnt(0);
        }
        __syncthreads();
        // -- pack: exp(k - Md) / copy v into packed-pair WMMA layouts --
#pragma unroll
        for (int e = 0; e < 8; ++e) {
            int flat = e * 256 + tid;       // 0..2047
            int buf  = flat >> 10;
            int rem  = flat & 1023;
            int isB  = rem >> 9;
            int j    = rem & 511;
            int dd   = j >> 4, p = j & 15;
            if (!isB) {
                float a0 = __expf((float)rawK[buf][2 * p + 0][dd] - Md[dd]);
                float a1 = __expf((float)rawK[buf][2 * p + 1][dd] - Md[dd]);
                Apack[buf][dd][p] = pack2h(a0, a1);
            } else {
                float b0 = (float)rawV[buf][2 * p + 0][dd];
                float b1 = (float)rawV[buf][2 * p + 1][dd];
                Bpack[buf][dd][p] = pack2h(b0, b1);
            }
        }
#else
        // fallback: gather straight from global (L2-resident)
#pragma unroll
        for (int e = 0; e < 8; ++e) {
            int flat = e * 256 + tid;
            int buf  = flat >> 10;
            int rem  = flat & 1023;
            int isB  = rem >> 9;
            int j    = rem & 511;
            int dd   = j >> 4, p = j & 15;
            int n    = (it * 2 + buf) * 32 + 2 * p;
            if (!isB) {
                float a0 = __expf((float)base[(size_t)n * OC + kcol0 + dd] - Md[dd]);
                float a1 = __expf((float)base[(size_t)(n + 1) * OC + kcol0 + dd] - Md[dd]);
                Apack[buf][dd][p] = pack2h(a0, a1);
            } else {
                float b0 = (float)base[(size_t)n * OC + vcol0 + dd];
                float b1 = (float)base[(size_t)(n + 1) * OC + vcol0 + dd];
                Bpack[buf][dd][p] = pack2h(b0, b1);
            }
        }
#endif
        __syncthreads();
        {
            FragH A, Bf;
#pragma unroll
            for (int v = 0; v < 8; ++v) {
                int kb = kbase(v, lhf);     // even
                A.u[v]  = Apack[kb_bf][kb_dt * 16 + col][kb >> 1];
                Bf.u[v] = Bpack[kb_bf][kb_et * 16 + col][kb >> 1];
            }
            accA = wmma16(A.h, Bf.h, accA);
        }
        __syncthreads();
    }

    // merge the two k-buffer partials per (d,e) tile with LDS float atomics
#pragma unroll
    for (int r = 0; r < 8; ++r) {
        int m = kb_dt * 16 + r + 8 * lhf;   // d
        int e = kb_et * 16 + col;
        atomicAdd(&attM[m][e], accA[r]);    // ds_add_f32
    }
    __syncthreads();

    // ---- scale rows by 1/Zd; repack as B fragment layout [e][d-pair] ----
    __shared__ unsigned attB[32][16];
    for (int i = tid; i < 512; i += 256) {
        int e = i >> 4, p = i & 15;
        float z0 = attM[2 * p + 0][e] / Zd[2 * p + 0];
        float z1 = attM[2 * p + 1][e] / Zd[2 * p + 1];
        attB[e][p] = pack2h(z0, z1);
    }
    __syncthreads();

    // ---- out_t[n][e] = q_t[n][:] @ att  (K = 32 = one WMMA step) ----
    FragH B0, B1;
#pragma unroll
    for (int v = 0; v < 8; ++v) {
        int kb = kbase(v, lhf);
        B0.u[v] = attB[0 * 16 + col][kb >> 1];
        B1.u[v] = attB[1 * 16 + col][kb >> 1];
    }

    __shared__ float xch[8][16][17];
    for (int mt = wave; mt < 256; mt += 8) {
        int n0 = mt * 16;
        const half_t* arow = base + (size_t)(n0 + col) * OC + qcol0;
        __builtin_prefetch(arow + 8 * OC, 0, 0);
        FragH A;
#pragma unroll
        for (int v = 0; v < 8; ++v)
            A.u[v] = *(const unsigned*)(arow + kbase(v, lhf));
        v8f c0 = wmma16(A.h, B0.h, zero8());
        v8f c1 = wmma16(A.h, B1.h, zero8());
#pragma unroll
        for (int t = 0; t < 2; ++t) {
            v8f cc = t ? c1 : c0;
#pragma unroll
            for (int r = 0; r < 8; ++r)
                xch[wave][r + 8 * lhf][col] = cc[r];
            const float* srcp = &xch[wave][col][lhf * 8];
            union { _Float16 h[8]; uint4 u; } pk;
#pragma unroll
            for (int e = 0; e < 8; ++e) pk.h[e] = (_Float16)srcp[e];
            *(uint4*)(outt + ((size_t)(b * NN) + n0 + col) * CC +
                      h * 32 + t * 16 + lhf * 8) = pk.u;
        }
    }
}

// ---------------------------------------------------------------------------
// Kernel 5: proj GEMM + bias.  out[b][o][n] = sum_c out_t[b][n][c]*Wp[o][c]+bp.
// D layout (lane = o column, VGPR r = consecutive n) -> 32B contiguous fp32
// stores straight into the channel-major output. No LDS needed.
// ---------------------------------------------------------------------------
__global__ __launch_bounds__(256) void k_proj(const half_t* __restrict__ outt,
                                              const half_t* __restrict__ wp,
                                              const float* __restrict__ bias,
                                              float* __restrict__ out) {
    int b    = blockIdx.x >> 8;
    int mt   = blockIdx.x & 255;
    int n0   = mt * 16;
    int tid  = threadIdx.x;
    int wave = tid >> 5;
    int lane = tid & 31;
    int col  = lane & 15;
    int lhf  = lane >> 4;

    const half_t* arow = outt + ((size_t)(b * NN) + n0 + col) * CC;
    v8f acc[2];
    acc[0] = zero8(); acc[1] = zero8();

    for (int ks = 0; ks < 8; ++ks) {
        int k0 = ks * 32;
        __builtin_prefetch(arow + k0 + 32, 0, 0);
        FragH A;
#pragma unroll
        for (int v = 0; v < 8; ++v)
            A.u[v] = *(const unsigned*)(arow + k0 + kbase(v, lhf));
#pragma unroll
        for (int t = 0; t < 2; ++t) {
            int o = (wave * 2 + t) * 16 + col;
            FragH Bf;
#pragma unroll
            for (int v = 0; v < 8; ++v)
                Bf.u[v] = *(const unsigned*)(wp + (size_t)o * CC + k0 + kbase(v, lhf));
            acc[t] = wmma16(A.h, Bf.h, acc[t]);
        }
    }

#pragma unroll
    for (int t = 0; t < 2; ++t) {
        int o = (wave * 2 + t) * 16 + col;
        float bv = bias[o];
        float* dst = out + ((size_t)(b * CC) + o) * NN + n0 + lhf * 8;
#pragma unroll
        for (int r = 0; r < 8; ++r) dst[r] = acc[t][r] + bv;
    }
}

// ---------------------------------------------------------------------------
extern "C" void kernel_launch(void* const* d_in, const int* in_sizes, int n_in,
                              void* d_out, int out_size, void* d_ws, size_t ws_size,
                              hipStream_t stream) {
    (void)in_sizes; (void)n_in; (void)out_size; (void)ws_size;
    const float* x     = (const float*)d_in[0];
    const float* gn_w  = (const float*)d_in[1];
    const float* gn_b  = (const float*)d_in[2];
    const float* wqkv  = (const float*)d_in[3];
    const float* wproj = (const float*)d_in[4];
    const float* bproj = (const float*)d_in[5];
    float* out = (float*)d_out;

    char* ws = (char*)d_ws;
    float*  stats = (float*)(ws + OFF_STATS);
    half_t* wq_h  = (half_t*)(ws + OFF_WQ);
    half_t* wp_h  = (half_t*)(ws + OFF_WP);
    half_t* xnt   = (half_t*)(ws + OFF_XNT);
    half_t* qkvt  = (half_t*)(ws + OFF_QKVT);
    half_t* outt  = (half_t*)(ws + OFF_OUTT);

    k_prep_w  <<<OC, 256, 0, stream>>>(wqkv, wproj, wq_h, wp_h);
    k_gn_stats<<<BB * NG, 256, 0, stream>>>(x, stats);
    k_gn_apply<<<BB * 64, 256, 0, stream>>>(x, stats, gn_w, gn_b, xnt);
    k_qkv     <<<BB * 256, 256, 0, stream>>>(xnt, wq_h, qkvt);
    k_att     <<<BB * HH, 256, 0, stream>>>(qkvt, outt);
    k_proj    <<<BB * 256, 256, 0, stream>>>(outt, wp_h, bproj, out);
}
